// AttentionHead_5583457485447
// MI455X (gfx1250) — compile-verified, matrix operand order checked by reading, not run on hardware
//
#include <hip/hip_runtime.h>

#define BB   8
#define NN   2048
#define DD   256
#define DKK  64
#define DVV  257
#define DVP  288          // DV padded to 9*32 = 18*16

// fragment constants
#define FR   512          // 32 lanes * 16 elements per fragment

typedef __attribute__((ext_vector_type(16))) __bf16 bf16x16;
typedef __attribute__((ext_vector_type(8)))  float  f32x8;

__device__ __forceinline__ unsigned short f2bf(float f) {
    unsigned int u = __builtin_bit_cast(unsigned int, f);
    u += 0x7FFFu + ((u >> 16) & 1u);   // round-to-nearest-even
    return (unsigned short)(u >> 16);
}

__device__ __forceinline__ f32x8 wmma_bf16(bf16x16 a, bf16x16 b, f32x8 c) {
    return __builtin_amdgcn_wmma_f32_16x16x32_bf16(false, a, false, b, (short)0, c, false, false);
}

// position (lane*16+e) of element (row-or-col mk, K) inside a 16x32 A-frag / 32x16 B-frag
__device__ __forceinline__ int frag_pos(int mk, int K) {
    int lane = (((K >> 3) & 1) << 4) | mk;
    int e    = ((K >> 4) << 3) | (K & 7);
    return lane * 16 + e;
}

__device__ __forceinline__ bf16x16 ldfrag(const unsigned short* p) {
    return *(const bf16x16*)p;
}

// ---------------- prep: X -> A-frag layout (bf16) ----------------
// xsw[b][it=128][kstep=8][512]
__global__ void prep_x(const float* __restrict__ X, unsigned short* __restrict__ xsw) {
    size_t idx = (size_t)blockIdx.x * 256 + threadIdx.x;
    if (idx >= (size_t)BB * NN * DD) return;
    int d  = idx % DD;
    size_t bn = idx / DD;
    int n  = bn % NN;
    int b  = bn / NN;
    int it = n >> 4, m = n & 15;
    int ks = d >> 5, kin = d & 31;
    size_t dst = (((size_t)(b * 128 + it) * 8 + ks) * FR) + frag_pos(m, kin);
    xsw[dst] = f2bf(X[idx]);
}

// ---------------- prep: Wq / Wk -> B-frag layout -----------------
// wsw[kstep=8][coltile=4][512]
__global__ void prep_w(const float* __restrict__ W, unsigned short* __restrict__ wsw) {
    int idx = blockIdx.x * 256 + threadIdx.x;
    if (idx >= DD * DKK) return;
    int c = idx % DKK, r = idx / DKK;
    int ks = r >> 5, kin = r & 31;
    int ct = c >> 4, nc = c & 15;
    wsw[((ks * 4 + ct) * FR) + frag_pos(nc, kin)] = f2bf(W[idx]);
}

// ---------------- prep: Wv (257x257 -> 288x288) B-frags ----------
// wvsw[a=9][wt=18][512]
__global__ void prep_wv(const float* __restrict__ Wv, unsigned short* __restrict__ wvsw) {
    int idx = blockIdx.x * 256 + threadIdx.x;
    if (idx >= DVP * DVP) return;
    int c = idx % DVP, r = idx / DVP;
    float v = (r < DVV && c < DVV) ? Wv[r * DVV + c] : 0.0f;
    int a = r >> 5, kin = r & 31;
    int wt = c >> 4, nc = c & 15;
    wvsw[((a * 18 + wt) * FR) + frag_pos(nc, kin)] = f2bf(v);
}

// ---------------- prep: Z (pad 257->288) B-frags -----------------
// zsw[b][jt=64][v=18][512]
__global__ void prep_z(const float* __restrict__ Z, unsigned short* __restrict__ zsw) {
    size_t idx = (size_t)blockIdx.x * 256 + threadIdx.x;
    if (idx >= (size_t)BB * NN * DVP) return;
    int col = idx % DVP;
    size_t bn = idx / DVP;
    int n = bn % NN, b = bn / NN;
    float v = (col < DVV) ? Z[((size_t)b * NN + n) * DVV + col] : 0.0f;
    int jt = n >> 5, kdim = n & 31;
    int vt = col >> 4, nc = col & 15;
    size_t dst = (((size_t)(b * 64 + jt) * 18 + vt) * FR) + frag_pos(nc, kdim);
    zsw[dst] = f2bf(v);
}

// -------- q/k projection: WMMA, outputs already fragment-swizzled --------
// grid (128, B, 8): z = 0..3 -> q col-tile z; z = 4..7 -> k col-tile z-4. block = 32
__global__ __launch_bounds__(32) void proj_qk(
    const unsigned short* __restrict__ xsw,
    const unsigned short* __restrict__ wqsw,
    const unsigned short* __restrict__ wksw,
    unsigned short* __restrict__ qsw,
    unsigned short* __restrict__ ksw)
{
    int it = blockIdx.x, b = blockIdx.y, z = blockIdx.z;
    int ct = z & 3;
    bool isK = (z >= 4);
    const unsigned short* wsw = isK ? wksw : wqsw;
    int lane = threadIdx.x;

    const unsigned short* xb = xsw + ((size_t)(b * 128 + it) * 8) * FR;
    f32x8 acc = {};
#pragma unroll
    for (int ks = 0; ks < 8; ++ks) {
        bf16x16 a  = ldfrag(xb + ks * FR + lane * 16);
        bf16x16 bm = ldfrag(wsw + (ks * 4 + ct) * FR + lane * 16);
        acc = wmma_bf16(a, bm, acc);
    }
    int half = lane >> 4, nc = lane & 15;
#pragma unroll
    for (int v = 0; v < 8; ++v) {
        int m  = v + 8 * half;     // token row within 16-tile
        int dk = ct * 16 + nc;     // 0..63
        unsigned short bits = f2bf(acc[v]);
        int f = dk >> 5, kin = dk & 31;
        if (!isK) {
            // q: A-frag layout  qsw[b][it][f=2][512]
            qsw[(((size_t)(b * 128 + it) * 2 + f) * FR) + frag_pos(m, kin)] = bits;
        } else {
            // k: B-frag layout  ksw[b][jt=64][frag=4][512]; token m is the key (column n)
            int jt = it >> 1, fragRow = it & 1;
            ksw[(((size_t)(b * 64 + jt) * 4 + fragRow * 2 + f) * FR) + frag_pos(m, kin)] = bits;
        }
    }
}

// ----------------------- fused attention main -----------------------
// grid (N/64, B), block 128 (4 waves); each wave owns one 16-query tile
__global__ __launch_bounds__(128) void attn_main(
    const unsigned short* __restrict__ qsw,
    const unsigned short* __restrict__ ksw,
    const unsigned short* __restrict__ zsw,
    const unsigned short* __restrict__ wvsw,
    const float* __restrict__ M,
    float* __restrict__ out)
{
    __shared__ __align__(64) unsigned short sS[4][FR];       // per-wave S re-swizzle (1 KB)
    __shared__ __align__(64) unsigned short sT[4][9 * FR];   // per-wave term re-swizzle (9 KB)

    const int b    = blockIdx.y;
    const int wave = threadIdx.x >> 5;
    const int lane = threadIdx.x & 31;
    const int it   = blockIdx.x * 4 + wave;   // 0..127
    const int i0   = it * 16;
    const int half = lane >> 4, nc = lane & 15;
    const float scale = 0.125f;               // 1/sqrt(64)

    // resident q fragments (16 x 64 over two K=32 frags)
    const unsigned short* qb = qsw + ((size_t)(b * 128 + it) * 2) * FR;
    bf16x16 qa0 = ldfrag(qb + 0 * FR + lane * 16);
    bf16x16 qa1 = ldfrag(qb + 1 * FR + lane * 16);

    f32x8 acc[18];
#pragma unroll
    for (int v = 0; v < 18; ++v) acc[v] = {};

    const float* Mbase = M + ((size_t)b * NN + i0) * NN;

#pragma unroll 1
    for (int jt = 0; jt < 64; ++jt) {
        const unsigned short* kb = ksw + ((size_t)(b * 64 + jt) * 4) * FR;
        bf16x16 k00 = ldfrag(kb + 0 * FR + lane * 16);   // keys 0..15, dk 0..31
        bf16x16 k01 = ldfrag(kb + 1 * FR + lane * 16);   // keys 0..15, dk 32..63
        bf16x16 k10 = ldfrag(kb + 2 * FR + lane * 16);   // keys 16..31, dk 0..31
        bf16x16 k11 = ldfrag(kb + 3 * FR + lane * 16);   // keys 16..31, dk 32..63

        f32x8 c0 = {}, c1 = {};
        c0 = wmma_bf16(qa0, k00, c0);
        c0 = wmma_bf16(qa1, k01, c0);
        c1 = wmma_bf16(qa0, k10, c1);
        c1 = wmma_bf16(qa1, k11, c1);

        // scale + relu + mask, re-swizzle C-layout -> A-frag layout via LDS
        const int j0 = jt * 32;
#pragma unroll
        for (int v = 0; v < 8; ++v) {
            int m = v + 8 * half;
            float s0 = c0[v] * scale;  s0 = s0 > 0.0f ? s0 : 0.0f;
            s0 *= Mbase[(size_t)m * NN + j0 + nc];
            sS[wave][frag_pos(m, nc)] = f2bf(s0);
            float s1 = c1[v] * scale;  s1 = s1 > 0.0f ? s1 : 0.0f;
            s1 *= Mbase[(size_t)m * NN + j0 + 16 + nc];
            sS[wave][frag_pos(m, 16 + nc)] = f2bf(s1);
        }
        // LDS ops are in-order within a wave (DScnt); compiler keeps the dependence.
        asm volatile("" ::: "memory");
        bf16x16 sa = ldfrag(&sS[wave][lane * 16]);

        const unsigned short* zb = zsw + ((size_t)(b * 64 + jt) * 18) * FR;
#pragma unroll
        for (int v = 0; v < 18; ++v) {
            bf16x16 zf = ldfrag(zb + v * FR + lane * 16);
            acc[v] = wmma_bf16(sa, zf, acc[v]);
        }
    }

    // dump term (16 x 288) as 9 A-frags into LDS
#pragma unroll
    for (int v = 0; v < 18; ++v) {
#pragma unroll
        for (int r = 0; r < 8; ++r) {
            int m = r + 8 * half;
            int K = v * 16 + nc;             // 0..287
            sT[wave][(K >> 5) * FR + frag_pos(m, K & 31)] = f2bf(acc[v][r]);
        }
    }
    asm volatile("" ::: "memory");

    // out = term @ Wv  (store only 257 valid columns)
    float* obase = out + ((size_t)b * NN + i0) * DVV;
#pragma unroll 1
    for (int wt = 0; wt < 18; ++wt) {
        f32x8 oc = {};
#pragma unroll
        for (int a = 0; a < 9; ++a) {
            bf16x16 ta = ldfrag(&sT[wave][a * FR + lane * 16]);
            bf16x16 wf = ldfrag(wvsw + ((size_t)(a * 18 + wt)) * FR + lane * 16);
            oc = wmma_bf16(ta, wf, oc);
        }
#pragma unroll
        for (int v = 0; v < 8; ++v) {
            int m = v + 8 * half;
            int w = wt * 16 + nc;
            if (w < DVV) obase[(size_t)m * DVV + w] = oc[v];
        }
    }
}

extern "C" void kernel_launch(void* const* d_in, const int* in_sizes, int n_in,
                              void* d_out, int out_size, void* d_ws, size_t ws_size,
                              hipStream_t stream) {
    (void)in_sizes; (void)n_in; (void)out_size; (void)ws_size;
    const float* Z  = (const float*)d_in[0];
    const float* X  = (const float*)d_in[1];
    const float* M  = (const float*)d_in[2];
    const float* Wq = (const float*)d_in[3];
    const float* Wk = (const float*)d_in[4];
    const float* Wv = (const float*)d_in[5];
    float* out = (float*)d_out;

    unsigned short* ws = (unsigned short*)d_ws;
    size_t o = 0;
    unsigned short* xsw  = ws + o; o += (size_t)BB * 128 * 8 * FR;   // 4.19M el
    unsigned short* wqsw = ws + o; o += (size_t)8 * 4 * FR;
    unsigned short* wksw = ws + o; o += (size_t)8 * 4 * FR;
    unsigned short* qsw  = ws + o; o += (size_t)BB * 128 * 2 * FR;
    unsigned short* ksw  = ws + o; o += (size_t)BB * 64 * 4 * FR;
    unsigned short* zsw  = ws + o; o += (size_t)BB * 64 * 18 * FR;
    unsigned short* wvsw = ws + o; o += (size_t)9 * 18 * FR;
    // total ~11.1M bf16 elements (~21.3 MB) of workspace

    prep_x <<<(BB * NN * DD + 255) / 256, 256, 0, stream>>>(X, xsw);
    prep_w <<<(DD * DKK + 255) / 256, 256, 0, stream>>>(Wq, wqsw);
    prep_w <<<(DD * DKK + 255) / 256, 256, 0, stream>>>(Wk, wksw);
    prep_wv<<<(DVP * DVP + 255) / 256, 256, 0, stream>>>(Wv, wvsw);
    prep_z <<<(BB * NN * DVP + 255) / 256, 256, 0, stream>>>(Z, zsw);

    proj_qk<<<dim3(128, BB, 8), 32, 0, stream>>>(xsw, wqsw, wksw, qsw, ksw);

    attn_main<<<dim3(NN / 64, BB), 128, 0, stream>>>(qsw, ksw, zsw, wvsw, M, out);
}
